// GAT_15178414424678
// MI455X (gfx1250) — matrix operand order repaired and checked by reading, not compiled, not run on hardware
//
#include <hip/hip_runtime.h>
#include <hip/hip_bf16.h>
#include <stdint.h>

// ---------------- constants (match reference setup_inputs) ----------------
#define NN 50000
#define NE 800000
#define NF 260
#define NH 4
#define NC 64
#define NHC 256
#define NB 256
#define NA 32

// GEMM block tile
#define BM 64
#define BN 64
#define BK 32
#define PADA 4
#define PADB 4

typedef __attribute__((ext_vector_type(16))) _Float16 v16h_t;
typedef __attribute__((ext_vector_type(8)))  float    v8f_t;

// ---------------- device helpers ----------------
__device__ __forceinline__ float selu_f(float x) {
    const float a = 1.6732632423543772f, s = 1.0507009873554805f;
    return x > 0.f ? s * x : s * a * (__expf(x) - 1.f);
}
// order-preserving float <-> uint encoding (for atomicMax on floats)
__device__ __forceinline__ unsigned fenc(float f) {
    unsigned u = __float_as_uint(f);
    return (u & 0x80000000u) ? ~u : (u | 0x80000000u);
}
__device__ __forceinline__ float fdec(unsigned u) {
    unsigned b = (u & 0x80000000u) ? (u ^ 0x80000000u) : ~u;
    return __uint_as_float(b);
}
__device__ __forceinline__ bool finite_f(float f) {
    return (__float_as_uint(f) & 0x7F800000u) != 0x7F800000u;
}

// ---------------- generic fill ----------------
__global__ void fill_u32(unsigned* __restrict__ p, unsigned v, size_t n) {
    size_t t = (size_t)blockIdx.x * blockDim.x + threadIdx.x;
    if (t < n) p[t] = v;
}

// ---------------- WMMA GEMM ----------------
// O = act(A(+Ab K-bias) @ W + Ob), f32 in/out, f16 WMMA math.
// 256 threads = 8 waves; block tile 64x64; each wave owns a 16x32 sub-tile
// (two v8f accumulators -> 2 v_wmma_f32_16x16x32_f16 per K-step).
// Fast unguarded staging path when the tile is fully in range; all call sites
// have K % 4 == 0 so float4 A staging is aligned.
template <int ACT, bool HAS_AB>
__global__ __launch_bounds__(256) void gemm_wmma(
    const float* __restrict__ A, const float* __restrict__ Ab,
    const float* __restrict__ W, const float* __restrict__ Ob,
    float* __restrict__ O, int M, int K, int Nc)
{
    __shared__ _Float16 sA[BM][BK + PADA];
    __shared__ _Float16 sB[BK][BN + PADB];
    const int tid  = threadIdx.x;
    const int lane = tid & 31;
    const int wave = tid >> 5;     // 0..7
    const int wy   = wave >> 1;    // 0..3 : 16-row band within 64
    const int wx   = wave & 1;     // 0..1 : 32-col band within 64
    const int m0 = blockIdx.y * BM;
    const int n0 = blockIdx.x * BN;

    v8f_t acc0 = {}, acc1 = {};

    // staging coordinates for this thread (2 chunks of A, 2 chunks of W per step)
    const int ar0 = tid >> 3, ac0 = (tid & 7) << 2;           // A chunk 0: rows 0..31
    const int ar1 = ar0 + 32;                                 // A chunk 1: rows 32..63
    const int br0 = tid >> 4, bc0 = (tid & 15) << 2;          // W chunk 0: rows 0..15
    const int br1 = br0 + 16;                                 // W chunk 1: rows 16..31

    for (int k0 = 0; k0 < K; k0 += BK) {
        const bool fullA = (m0 + BM <= M) && (k0 + BK <= K);
        const bool fullB = (n0 + BN <= Nc) && (k0 + BK <= K);

        if (fullA) {
            // unguarded: float4 load -> cvt_pk -> ds_store_b64
            #pragma unroll
            for (int q = 0; q < 2; ++q) {
                int r = q ? ar1 : ar0;
                float4 v = *(const float4*)(A + (size_t)(m0 + r) * K + k0 + ac0);
                if (HAS_AB) {
                    float4 b = *(const float4*)(Ab + k0 + ac0);
                    v.x += b.x; v.y += b.y; v.z += b.z; v.w += b.w;
                }
                _Float16* p = &sA[r][ac0];
                p[0] = (_Float16)v.x; p[1] = (_Float16)v.y;
                p[2] = (_Float16)v.z; p[3] = (_Float16)v.w;
            }
        } else {
            #pragma unroll
            for (int q = 0; q < 2; ++q) {
                int r = q ? ar1 : ar0;
                int gr = m0 + r;
                float4 v = make_float4(0.f, 0.f, 0.f, 0.f);
                if (gr < M) {
                    #pragma unroll
                    for (int i = 0; i < 4; ++i) {
                        int g = k0 + ac0 + i;
                        if (g < K) {
                            float f = A[(size_t)gr * K + g];
                            if (HAS_AB) f += Ab[g];
                            (&v.x)[i] = f;
                        }
                    }
                }
                _Float16* p = &sA[r][ac0];
                p[0] = (_Float16)v.x; p[1] = (_Float16)v.y;
                p[2] = (_Float16)v.z; p[3] = (_Float16)v.w;
            }
        }

        if (fullB) {
            #pragma unroll
            for (int q = 0; q < 2; ++q) {
                int r = q ? br1 : br0;
                float4 v = *(const float4*)(W + (size_t)(k0 + r) * Nc + n0 + bc0);
                _Float16* p = &sB[r][bc0];
                p[0] = (_Float16)v.x; p[1] = (_Float16)v.y;
                p[2] = (_Float16)v.z; p[3] = (_Float16)v.w;
            }
        } else {
            #pragma unroll
            for (int q = 0; q < 2; ++q) {
                int r = q ? br1 : br0;
                int gk = k0 + r;
                float4 v = make_float4(0.f, 0.f, 0.f, 0.f);
                if (gk < K) {
                    #pragma unroll
                    for (int i = 0; i < 4; ++i)
                        if (n0 + bc0 + i < Nc) (&v.x)[i] = W[(size_t)gk * Nc + n0 + bc0 + i];
                }
                _Float16* p = &sB[r][bc0];
                p[0] = (_Float16)v.x; p[1] = (_Float16)v.y;
                p[2] = (_Float16)v.z; p[3] = (_Float16)v.w;
            }
        }
        __syncthreads();

        // ---- fragments per CDNA5 16-bit WMMA VGPR layouts (ISA 7.12.2) ----
        v16h_t af, bf0, bf1;
        const int mrow = (lane & 15) + wy * 16;
        const int kloA = (lane & 16) ? 8 : 0;    // A VGPR 0..3 K-base
        const int khiA = (lane & 16) ? 24 : 16;  // A VGPR 4..7 K-base
        const int nb   = wx * 32;
        #pragma unroll
        for (int h = 0; h < 16; ++h) {
            int v = h >> 1, p = h & 1;
            int kk = (v < 4) ? (kloA + 2 * v + p) : (khiA + 2 * (v - 4) + p);
            af[h]  = sA[mrow][kk];
            bf0[h] = sB[kk][nb + (lane & 15)];        // B: col N = lane&15, K in same order as A
            bf1[h] = sB[kk][nb + 16 + (lane & 15)];
        }
        acc0 = __builtin_amdgcn_wmma_f32_16x16x32_f16(
            false, af, false, bf0, (short)0, acc0, false, false);
        acc1 = __builtin_amdgcn_wmma_f32_16x16x32_f16(
            false, af, false, bf1, (short)0, acc1, false, false);
        __syncthreads();
    }

    // ---- store: C/D layout: VGPR v -> M = v (+8 for lanes 16..31), N = lane&15 ----
    const int col = lane & 15;
    const int rb  = (lane & 16) ? 8 : 0;
    const int gc0 = n0 + wx * 32 + col;
    const int gc1 = gc0 + 16;
    const bool fullO = (m0 + BM <= M) && (n0 + BN <= Nc);
    if (fullO) {
        float ob0 = Ob[gc0], ob1 = Ob[gc1];
        #pragma unroll
        for (int v = 0; v < 8; ++v) {
            int gr = m0 + wy * 16 + rb + v;
            float o0 = acc0[v] + ob0;
            float o1 = acc1[v] + ob1;
            if (ACT == 1) { o0 = selu_f(o0); o1 = selu_f(o1); }
            O[(size_t)gr * Nc + gc0] = o0;
            O[(size_t)gr * Nc + gc1] = o1;
        }
    } else {
        #pragma unroll
        for (int v = 0; v < 8; ++v) {
            int gr = m0 + wy * 16 + rb + v;
            if (gr < M) {
                if (gc0 < Nc) {
                    float o = acc0[v] + Ob[gc0];
                    if (ACT == 1) o = selu_f(o);
                    O[(size_t)gr * Nc + gc0] = o;
                }
                if (gc1 < Nc) {
                    float o = acc1[v] + Ob[gc1];
                    if (ACT == 1) o = selu_f(o);
                    O[(size_t)gr * Nc + gc1] = o;
                }
            }
        }
    }
}

// ---------------- GATv2 edge pass A: scores + segment max ----------------
// thread per (edge, head): e = sum_c leakyrelu(xl[src]+xr[dst]) * att
__global__ void edge_logits(const float* __restrict__ xl, const float* __restrict__ xr,
                            const float* __restrict__ att,
                            const int* __restrict__ src, const int* __restrict__ dst,
                            float* __restrict__ e, unsigned* __restrict__ emax)
{
    int t = blockIdx.x * blockDim.x + threadIdx.x;
    if (t >= NE * NH) return;
    int edge = t >> 2, h = t & 3;
    int s = src[edge], d = dst[edge];
    const float4* pl = (const float4*)(xl + (size_t)s * NHC + h * NC);
    const float4* pr = (const float4*)(xr + (size_t)d * NHC + h * NC);
    const float4* pa = (const float4*)(att + h * NC);
    float acc = 0.f;
    #pragma unroll 4
    for (int i = 0; i < NC / 4; ++i) {
        float4 a = pl[i], b = pr[i], w = pa[i];
        float m;
        m = a.x + b.x; acc += (m > 0.f ? m : 0.2f * m) * w.x;
        m = a.y + b.y; acc += (m > 0.f ? m : 0.2f * m) * w.y;
        m = a.z + b.z; acc += (m > 0.f ? m : 0.2f * m) * w.z;
        m = a.w + b.w; acc += (m > 0.f ? m : 0.2f * m) * w.w;
    }
    e[t] = acc;
    atomicMax(emax + (size_t)d * NH + h, fenc(acc));
}

// ---------------- edge pass B: ea = exp(e - emax[dst]); denom += ea ----------------
__global__ void edge_expsum(const float* __restrict__ e, const unsigned* __restrict__ emax,
                            const int* __restrict__ dst,
                            float* __restrict__ ea, float* __restrict__ den)
{
    int t = blockIdx.x * blockDim.x + threadIdx.x;
    if (t >= NE * NH) return;
    int edge = t >> 2, h = t & 3;
    int d = dst[edge];
    float m = fdec(emax[(size_t)d * NH + h]);
    if (!finite_f(m)) m = 0.f;
    float v = __expf(e[t] - m);
    ea[t] = v;
    atomicAdd(den + (size_t)d * NH + h, v);
}

// ---------------- edge pass C: out[dst] += xl[src] * alpha ----------------
// 64 threads per edge, each does a float4 slice of one head
__global__ void edge_aggregate(const float* __restrict__ xl, const float* __restrict__ ea,
                               const float* __restrict__ den,
                               const int* __restrict__ src, const int* __restrict__ dst,
                               float* __restrict__ out)
{
    int t = blockIdx.x * blockDim.x + threadIdx.x;
    if (t >= NE * 64) return;
    int edge = t >> 6;
    int j = t & 63;
    int h = j >> 4;
    int c0 = (j & 15) << 2;
    int s = src[edge], d = dst[edge];
    float alpha = ea[(size_t)edge * NH + h] / (den[(size_t)d * NH + h] + 1e-16f);
    float4 x = *(const float4*)(xl + (size_t)s * NHC + h * NC + c0);
    float* o = out + (size_t)d * NHC + h * NC + c0;
    atomicAdd(o + 0, x.x * alpha);
    atomicAdd(o + 1, x.y * alpha);
    atomicAdd(o + 2, x.z * alpha);
    atomicAdd(o + 3, x.w * alpha);
}

// ---------------- pooling ----------------
__global__ void pool_acc(const float* __restrict__ hbuf, const int* __restrict__ batch,
                         float* __restrict__ psum, unsigned* __restrict__ pmax,
                         float* __restrict__ cnt)
{
    int t = blockIdx.x * blockDim.x + threadIdx.x;
    if (t >= NN * NC) return;
    int n = t >> 6, c = t & 63;
    int b = batch[n];
    float v = hbuf[t];
    atomicAdd(psum + (size_t)b * NC + c, v);
    atomicMax(pmax + (size_t)b * NC + c, fenc(v));
    if (c == 0) atomicAdd(cnt + b, 1.0f);
}

__global__ void pool_fin(const float* __restrict__ psum, const unsigned* __restrict__ pmax,
                         const float* __restrict__ cnt, float* __restrict__ g, int off)
{
    int t = blockIdx.x * blockDim.x + threadIdx.x;
    if (t >= NB * NC) return;
    int b = t >> 6, c = t & 63;
    float n = fmaxf(cnt[b], 1.0f);
    g[(size_t)b * 256 + off + c]      = psum[t] / n;
    g[(size_t)b * 256 + off + NC + c] = fdec(pmax[t]);
}

// ---------------- final: log_softmax, gather by action, entropy, value ------------
__global__ __launch_bounds__(32) void heads_out(const float* __restrict__ logits,
                                                const float* __restrict__ value,
                                                const int* __restrict__ action,
                                                float* __restrict__ out)
{
    int b = blockIdx.x;
    int lane = threadIdx.x;
    float l = logits[(size_t)b * NA + lane];
    float m = l;
    for (int o = 16; o; o >>= 1) m = fmaxf(m, __shfl_xor(m, o, 32));
    float ex = __expf(l - m);
    float s = ex;
    for (int o = 16; o; o >>= 1) s += __shfl_xor(s, o, 32);
    float lse = m + logf(s);
    float lp = l - lse;
    float ent = -__expf(lp) * lp;
    for (int o = 16; o; o >>= 1) ent += __shfl_xor(ent, o, 32);
    int a = action[b];
    float lpa = __shfl(lp, a, 32);
    if (lane == 0) {
        out[b]            = (float)a;     // action
        out[NB + b]       = lpa;          // logp
        out[2 * NB + b]   = ent;          // entropy
        out[3 * NB + b]   = value[b];     // value
    }
}

// ---------------- host-side orchestration ----------------
static inline void launch_gemm(hipStream_t st, const float* A, const float* Ab,
                               const float* W, const float* Ob, float* O,
                               int M, int K, int Nc, int act)
{
    dim3 g((Nc + BN - 1) / BN, (M + BM - 1) / BM);
    if (Ab) {
        if (act) gemm_wmma<1, true><<<g, dim3(256), 0, st>>>(A, Ab, W, Ob, O, M, K, Nc);
        else     gemm_wmma<0, true><<<g, dim3(256), 0, st>>>(A, Ab, W, Ob, O, M, K, Nc);
    } else {
        if (act) gemm_wmma<1, false><<<g, dim3(256), 0, st>>>(A, Ab, W, Ob, O, M, K, Nc);
        else     gemm_wmma<0, false><<<g, dim3(256), 0, st>>>(A, Ab, W, Ob, O, M, K, Nc);
    }
}
static inline void launch_fill(hipStream_t st, void* p, unsigned v, size_t n)
{
    fill_u32<<<(unsigned)((n + 255) / 256), 256, 0, st>>>((unsigned*)p, v, n);
}

extern "C" void kernel_launch(void* const* d_in, const int* in_sizes, int n_in,
                              void* d_out, int out_size, void* d_ws, size_t ws_size,
                              hipStream_t stream)
{
    (void)in_sizes; (void)n_in; (void)out_size; (void)ws_size;
    // params (dict order)
    const float* c1_Wl  = (const float*)d_in[0];
    const float* c1_bl  = (const float*)d_in[1];
    const float* c1_Wr  = (const float*)d_in[2];
    const float* c1_br  = (const float*)d_in[3];
    const float* c1_att = (const float*)d_in[4];
    const float* c1_bias= (const float*)d_in[5];
    const float* l1_W   = (const float*)d_in[6];
    const float* l1_b   = (const float*)d_in[7];
    const float* c2_Wl  = (const float*)d_in[8];
    const float* c2_bl  = (const float*)d_in[9];
    const float* c2_Wr  = (const float*)d_in[10];
    const float* c2_br  = (const float*)d_in[11];
    const float* c2_att = (const float*)d_in[12];
    const float* c2_bias= (const float*)d_in[13];
    const float* l2_W   = (const float*)d_in[14];
    const float* l2_b   = (const float*)d_in[15];
    const float* sum_W  = (const float*)d_in[16];
    const float* sum_b  = (const float*)d_in[17];
    const float* sh_W   = (const float*)d_in[18];
    const float* sh_b   = (const float*)d_in[19];
    const float* pi0_W  = (const float*)d_in[20];
    const float* pi0_b  = (const float*)d_in[21];
    const float* pi1_W  = (const float*)d_in[22];
    const float* pi1_b  = (const float*)d_in[23];
    const float* pi2_W  = (const float*)d_in[24];
    const float* pi2_b  = (const float*)d_in[25];
    const float* v0_W   = (const float*)d_in[26];
    const float* v0_b   = (const float*)d_in[27];
    const float* v1_W   = (const float*)d_in[28];
    const float* v1_b   = (const float*)d_in[29];
    const float* v2_W   = (const float*)d_in[30];
    const float* v2_b   = (const float*)d_in[31];
    const float* x      = (const float*)d_in[32];
    const int*   ei     = (const int*)  d_in[33];
    const int*   batch  = (const int*)  d_in[34];
    const int*   action = (const int*)  d_in[35];
    const int* src = ei;
    const int* dst = ei + NE;

    // workspace carve (floats)
    float* wf = (float*)d_ws;
    size_t off = 0;
    auto takef = [&](size_t n) { float* p = wf + off; off += n; return p; };
    float*    XL   = takef((size_t)NN * NHC);    // lin_l output (both layers)
    float*    XR   = takef((size_t)NN * NHC);    // lin_r output; reused as GAT out accumulator
    float*    EB   = takef((size_t)NE * NH);     // e then ea (in-place)
    unsigned* EMAX = (unsigned*)takef((size_t)NN * NH);
    float*    DEN  = takef((size_t)NN * NH);
    float*    HB   = takef((size_t)NN * NC);     // h after l1 / l2
    float*    G    = takef((size_t)NB * 256);    // concat(g1, g2)
    float*    PSUM = takef((size_t)NB * NC);
    unsigned* PMAX = (unsigned*)takef((size_t)NB * NC);
    float*    CNT  = takef(NB);
    float*    T1   = takef((size_t)NB * 128);
    float*    T2   = takef((size_t)NB * 64);
    float*    T3   = takef((size_t)NB * 64);
    float*    T4   = takef((size_t)NB * 64);
    float*    TLOG = takef((size_t)NB * NA);
    float*    TVAL = takef((size_t)NB * 16);

    const unsigned ENC_NEG_INF = 0x007FFFFFu; // fenc(-inf)
    const int TB = 256;
    const int gEH  = (NE * NH + TB - 1) / TB;
    const int gE64 = (NE * 64 + TB - 1) / TB;

    auto gat_edges = [&](const float* att) {
        launch_fill(stream, EMAX, ENC_NEG_INF, (size_t)NN * NH);
        edge_logits<<<gEH, TB, 0, stream>>>(XL, XR, att, src, dst, EB, EMAX);
        launch_fill(stream, DEN, 0u, (size_t)NN * NH);
        edge_expsum<<<gEH, TB, 0, stream>>>(EB, EMAX, dst, EB, DEN);
        launch_fill(stream, XR, 0u, (size_t)NN * NHC);   // XR becomes GAT output
        edge_aggregate<<<gE64, TB, 0, stream>>>(XL, EB, DEN, src, dst, XR);
    };
    auto pool = [&](const float* h, int goff) {
        launch_fill(stream, PSUM, 0u, (size_t)NB * NC);
        launch_fill(stream, PMAX, ENC_NEG_INF, (size_t)NB * NC);
        launch_fill(stream, CNT, 0u, NB);
        pool_acc<<<(NN * NC + TB - 1) / TB, TB, 0, stream>>>(h, batch, PSUM, PMAX, CNT);
        pool_fin<<<(NB * NC + TB - 1) / TB, TB, 0, stream>>>(PSUM, PMAX, CNT, G, goff);
    };

    // ---- layer 1 ----
    launch_gemm(stream, x, nullptr, c1_Wl, c1_bl, XL, NN, NF, NHC, 0);
    launch_gemm(stream, x, nullptr, c1_Wr, c1_br, XR, NN, NF, NHC, 0);
    gat_edges(c1_att);
    launch_gemm(stream, XR, c1_bias, l1_W, l1_b, HB, NN, NHC, NC, 1); // selu
    pool(HB, 0);

    // ---- layer 2 ----
    launch_gemm(stream, HB, nullptr, c2_Wl, c2_bl, XL, NN, NC, NHC, 0);
    launch_gemm(stream, HB, nullptr, c2_Wr, c2_br, XR, NN, NC, NHC, 0);
    gat_edges(c2_att);
    launch_gemm(stream, XR, c2_bias, l2_W, l2_b, HB, NN, NHC, NC, 1); // selu
    pool(HB, 128);

    // ---- heads ----
    launch_gemm(stream, G,  nullptr, sum_W, sum_b, T1, NB, 256, 128, 0);
    launch_gemm(stream, T1, nullptr, sh_W,  sh_b,  T2, NB, 128, 64, 1);
    launch_gemm(stream, T2, nullptr, pi0_W, pi0_b, T3, NB, 64, 64, 1);
    launch_gemm(stream, T3, nullptr, pi1_W, pi1_b, T4, NB, 64, 64, 1);
    launch_gemm(stream, T4, nullptr, pi2_W, pi2_b, TLOG, NB, 64, NA, 0);
    launch_gemm(stream, T2, nullptr, v0_W, v0_b, T3, NB, 64, 64, 1);
    launch_gemm(stream, T3, nullptr, v1_W, v1_b, T4, NB, 64, 64, 1);
    launch_gemm(stream, T4, nullptr, v2_W, v2_b, TVAL, NB, 64, 1, 0);

    heads_out<<<NB, 32, 0, stream>>>(TLOG, TVAL, action, (float*)d_out);
}
